// BoundaryAwareLoss_79121887527331
// MI455X (gfx1250) — compile-verified, matrix operand order checked by reading, not run on hardware
//
#include <hip/hip_runtime.h>
#include <hip/hip_bf16.h>
#include <stdint.h>
#include <stddef.h>

// Problem constants (from reference setup_inputs / DILATION_RATIO)
#define B_    8
#define H_    544
#define W_    544
#define PAD_  7            // k = int(0.02*sqrt(2)*544) = 15 -> pad 7
#define KTAP_ 15
#define HW_   (H_ * W_)
#define NGROUPS_ (HW_ / 16)   // 18496 groups of 16 pixels (W % 16 == 0 -> row aligned)
#define NTHR  256
#define NBLK_FUSED 512

typedef __attribute__((ext_vector_type(16))) _Float16 v16h;
typedef __attribute__((ext_vector_type(8)))  float    v8f;

// clang-22 prototype: __builtin_amdgcn_global_load_async_to_lds_b32(as1 int*, as3 int*, Ii, Ii)
#define AS1I(p) ((__attribute__((address_space(1))) int*)(p))
#define AS3I(p) ((__attribute__((address_space(3))) int*)(p))

// ---------------------------------------------------------------------------
// Kernel A: horizontal 15-tap max/min over the binary target. One block per
// image row h; all 8 batches' rows (8*544 floats = 17 KB) staged into LDS via
// CDNA5 async global->LDS copies (ASYNCcnt path). Fixed 15-tap window with
// edge-clamped taps (binary max/min idempotent => clamp == exclude-OOB, same
// as reduce_window's -inf/+inf padding). Output: one uint16 per pixel packing
// all 8 batches: bit 2i = dilated_i, bit 2i+1 = eroded_i.
// ---------------------------------------------------------------------------
__global__ __launch_bounds__(NTHR)
void row_minmax_kernel(const float* __restrict__ target,
                       unsigned short* __restrict__ mask16) {
    __shared__ float rows[B_][W_];                 // 17408 bytes
    const int h = blockIdx.x;

    // B_*W_ = 4352 = 17 * 256: uniform staging loop, one async dword per lane
    for (int idx = threadIdx.x; idx < B_ * W_; idx += NTHR) {
        const int b = idx / W_;
        const int w = idx - b * W_;
        const float* src = target + ((size_t)b * H_ + h) * W_ + w;
#if __has_builtin(__builtin_amdgcn_global_load_async_to_lds_b32)
        __builtin_amdgcn_global_load_async_to_lds_b32(
            AS1I(const_cast<float*>(src)), AS3I(&rows[b][w]), 0, 0);
#else
        rows[b][w] = *src;
#endif
    }
#if __has_builtin(__builtin_amdgcn_s_wait_asynccnt)
    __builtin_amdgcn_s_wait_asynccnt(0);
#else
    asm volatile("s_wait_asynccnt 0" ::: "memory");
#endif
    __syncthreads();

    for (int w = threadIdx.x; w < W_; w += NTHR) {
        int pack = 0;
#pragma unroll
        for (int b = 0; b < B_; ++b) {
            float mx = 0.0f, mn = 1.0f;
#pragma unroll
            for (int d = -PAD_; d <= PAD_; ++d) {
                int x = w + d;
                x = x < 0 ? 0 : (x > W_ - 1 ? W_ - 1 : x);
                const float v = rows[b][x];
                mx = fmaxf(mx, v);
                mn = fminf(mn, v);
            }
            pack |= ((int)(mx > 0.5f)) << (2 * b);
            pack |= ((int)(mn > 0.5f)) << (2 * b + 1);
        }
        mask16[(size_t)h * W_ + w] = (unsigned short)pack;
    }
}

// ---------------------------------------------------------------------------
// Kernel B: fused vertical pass + boundary count via WMMA + BCE + dice sums.
// Each wave handles 16 consecutive pixels of one image row per iteration:
//   lane (l&15) -> pixel, (l>>4) -> batch/j half {0..3} vs {4..7}.
// Vertical pass: 15 independent u16 loads (batched, then combined) give all
// 8 batches at once: OR-accumulate dilate bits, AND-accumulate erode bits.
// WMMA: A holds boundary bits (exact in f16), B = all-ones
//   =>  D[m][*] = sum_k A[m][k] = count(pixel m)  (robust to A K-slot layout).
// Lanes 0/16 own C rows 0..7 / 8..15 and publish counts through LDS
// (ds_store_b128, wavefront-fenced; same-wave DS ops are in-order).
// Deterministic per-block tree reduction, no atomics.
// ---------------------------------------------------------------------------
__global__ __launch_bounds__(NTHR)
void fused_loss_kernel(const float* __restrict__ pred,
                       const float* __restrict__ target,
                       const unsigned short* __restrict__ mask16,
                       float* __restrict__ partials) {
    __shared__ float counts[NTHR / 32][16];
    __shared__ float redsm[4][NTHR];

    const int tid  = threadIdx.x;
    const int lane = tid & 31;
    const int wv   = tid >> 5;
    const int px16 = lane & 15;
    const int half = lane >> 4;
    const int gw0  = blockIdx.x * (NTHR / 32) + wv;
    const int TW   = gridDim.x * (NTHR / 32);

    float sp = 0.0f, st = 0.0f, spt = 0.0f, sci = 0.0f;

    // loop-invariant WMMA operands: all-ones B, zero template for A
    v16h bones, azero;
#pragma unroll
    for (int e = 0; e < 16; ++e) { bones[e] = (_Float16)1.0f; azero[e] = (_Float16)0.0f; }

    for (int g = gw0; g < NGROUPS_; g += TW) {        // wave-uniform -> EXEC all 1s
        const int base = g * 16;
        const int h = base / W_;
        const int w = (base % W_) + px16;

        // ---- vertical 15-tap window over packed uint16 plane (L2-resident).
        // Issue all taps first so they can go out as one load clause.
        unsigned int tap[KTAP_];
#pragma unroll
        for (int d = 0; d < KTAP_; ++d) {
            int hh = h + d - PAD_;
            hh = hh < 0 ? 0 : (hh > H_ - 1 ? H_ - 1 : hh);
            tap[d] = mask16[(size_t)hh * W_ + w];
        }
        unsigned int ob = 0, ab = 0xFFFFu;
#pragma unroll
        for (int d = 0; d < KTAP_; ++d) { ob |= tap[d]; ab &= tap[d]; }

        // this lane's 4 batches: boundary_i = dilated_i - eroded_i  in {0,1}
        float bnd[4];
#pragma unroll
        for (int q = 0; q < 4; ++q) {
            const int i = half * 4 + q;
            const int dil = (int)((ob >> (2 * i)) & 1u);
            const int ero = (int)((ab >> (2 * i + 1)) & 1u);
            bnd[q] = (float)(dil - ero);
        }

        // ---- count(pixel) = sum_i boundary_i via matrix engine ----
        v16h a = azero;
#pragma unroll
        for (int q = 0; q < 4; ++q) a[q] = (_Float16)bnd[q];
        v8f c;
#pragma unroll
        for (int e = 0; e < 8; ++e) c[e] = 0.0f;
        c = __builtin_amdgcn_wmma_f32_16x16x32_f16(
                /*neg_a=*/false, a, /*neg_b=*/false, bones,
                /*c_mod=*/(short)0, c, /*reuse_a=*/false, /*reuse_b=*/false);

        if (px16 == 0) {                             // lane 0 -> rows 0..7, lane 16 -> 8..15
#pragma unroll
            for (int r = 0; r < 8; ++r) counts[wv][half * 8 + r] = c[r];
        }
        __builtin_amdgcn_fence(__ATOMIC_RELEASE, "wavefront");
        __builtin_amdgcn_wave_barrier();             // same-wave DS ops are in-order
        __builtin_amdgcn_fence(__ATOMIC_ACQUIRE, "wavefront");
        const float mycount = counts[wv][px16];

        // ---- BCE inner sum over this lane's 4 batches + dice partials ----
        float inner = 0.0f;
#pragma unroll
        for (int q = 0; q < 4; ++q) {
            const int j = half * 4 + q;
            const size_t off = ((size_t)j * H_ + h) * W_ + w;
            const float p = pred[off];
            const float t = target[off];
            sp  += p;
            st  += t;
            spt += p * t;
            const float s   = 1.0f / (1.0f + __expf(-p));       // sigmoid
            const float lp  = fmaxf(__logf(s),        -100.0f); // clamp like torch BCE
            const float l1p = fmaxf(__logf(1.0f - s), -100.0f);
            inner += t * lp + (1.0f - t) * l1p;
        }
        sci += mycount * inner;
    }

    // ---- deterministic block tree reduction ----
    redsm[0][tid] = sp; redsm[1][tid] = st; redsm[2][tid] = spt; redsm[3][tid] = sci;
    __syncthreads();
    for (int s = NTHR / 2; s > 0; s >>= 1) {
        if (tid < s) {
#pragma unroll
            for (int k = 0; k < 4; ++k) redsm[k][tid] += redsm[k][tid + s];
        }
        __syncthreads();
    }
    if (tid == 0) {
#pragma unroll
        for (int k = 0; k < 4; ++k) partials[blockIdx.x * 4 + k] = redsm[k][0];
    }
}

// ---------------------------------------------------------------------------
// Kernel C: single-block deterministic final reduction + scalar assembly.
// ---------------------------------------------------------------------------
__global__ __launch_bounds__(NTHR)
void finalize_kernel(const float* __restrict__ partials, int nblk,
                     float* __restrict__ out) {
    __shared__ float redsm[4][NTHR];
    const int tid = threadIdx.x;
    float s[4] = {0.0f, 0.0f, 0.0f, 0.0f};
    for (int i = tid; i < nblk; i += NTHR) {
#pragma unroll
        for (int k = 0; k < 4; ++k) s[k] += partials[i * 4 + k];
    }
#pragma unroll
    for (int k = 0; k < 4; ++k) redsm[k][tid] = s[k];
    __syncthreads();
    for (int sdx = NTHR / 2; sdx > 0; sdx >>= 1) {
        if (tid < sdx) {
#pragma unroll
            for (int k = 0; k < 4; ++k) redsm[k][tid] += redsm[k][tid + sdx];
        }
        __syncthreads();
    }
    if (tid == 0) {
        const float sum_p  = redsm[0][0];
        const float sum_t  = redsm[1][0];
        const float sum_pt = redsm[2][0];
        const float sum_ci = redsm[3][0];
        const float dice = 1.0f - (2.0f * sum_pt + 1.0f) / (sum_p + sum_t + 1.0f);
        const float denom = (float)((long long)B_ * B_ * H_ * W_);   // mean over (B,B,H,W)
        const float bce = -sum_ci / denom;
        out[0] = dice + bce;   // ALPHA = BETA = 1
    }
}

// ---------------------------------------------------------------------------
extern "C" void kernel_launch(void* const* d_in, const int* in_sizes, int n_in,
                              void* d_out, int out_size, void* d_ws, size_t ws_size,
                              hipStream_t stream) {
    const float* pred   = (const float*)d_in[0];
    const float* target = (const float*)d_in[1];

    // workspace layout: [mask16: HW u16 = 591872 B][partials: NBLK*4 f32]
    unsigned short* mask16 = (unsigned short*)d_ws;
    float* partials        = (float*)((unsigned char*)d_ws + (size_t)HW_ * 2);

    row_minmax_kernel<<<H_, NTHR, 0, stream>>>(target, mask16);
    fused_loss_kernel<<<NBLK_FUSED, NTHR, 0, stream>>>(pred, target, mask16, partials);
    finalize_kernel<<<1, NTHR, 0, stream>>>(partials, NBLK_FUSED, (float*)d_out);
}
